// SutraV053_73993696575907
// MI455X (gfx1250) — matrix operand order repaired and checked by reading, not compiled, not run on hardware
//
#include <hip/hip_runtime.h>
#include <cmath>

// ---- problem constants (match reference) ----
#define BD 4
#define TD 512
#define VD 50257
#define DD 768
#define FFD 1536
#define NS 5
#define NSL 8
#define WIN 4
#define NSTEPS 8
#define MROWS (BD*TD)            // 2048 tokens
#define RSQRT_D 0.03608439182435161f   // 1/sqrt(768)
#define KS_D  (DD/32)            // 24 k-steps for K=768
#define KS_FF (FFD/32)           // 48 k-steps for K=1536

// ---- WMMA types (gfx1250, wave32) ----
typedef __attribute__((ext_vector_type(16))) _Float16 v16h;
typedef __attribute__((ext_vector_type(8)))  float    v8f;

union CFrag { v8f v; float f[8]; };

// ---- async global->LDS path (CDNA5), guarded so the file always compiles ----
#if defined(__AMDGCN__) && __has_builtin(__builtin_amdgcn_global_load_async_to_lds_b128) && __has_builtin(__builtin_amdgcn_s_wait_asynccnt)
#define USE_ASYNC_LDS 1
#endif

#ifdef USE_ASYNC_LDS
typedef int v4i_ty __attribute__((vector_size(16)));
__device__ __forceinline__ void async_cp16(const _Float16* g, _Float16* l) {
  __builtin_amdgcn_global_load_async_to_lds_b128(
      (__attribute__((address_space(1))) v4i_ty*)g,
      (__attribute__((address_space(3))) v4i_ty*)l,
      0, 0);
}
__device__ __forceinline__ void async_wait0() { __builtin_amdgcn_s_wait_asynccnt(0); }
#else
__device__ __forceinline__ void async_wait0() {}
#endif

// =====================================================================
// Fragment-panel layout.
// A-side panel (activations, M x K): for 16-row tile mt, 32-K step ks:
//   panel = A16[((mt*KS + ks)*32 + lane)*16 + idx], lane/idx per the
//   16-bit A operand layout of v_wmma_f32_16x16x32_f16.
// B-side panel (weights, K x N): same shape per (nt, ks) with the B
//   operand layout. Each lane's fragment = 32 contiguous bytes.
// =====================================================================
__device__ __forceinline__ void storeApanel(_Float16* __restrict__ P, int KS,
                                            int row, int col, float v) {
  int mt = row >> 4, lr = row & 15;
  int ks = col >> 5, kkl = col & 31;
  int gg  = (kkl >> 3) & 1;
  int idx = (kkl & 7) + ((kkl >= 16) ? 8 : 0);
  int lane = lr + (gg << 4);
  P[((((size_t)mt * KS + ks) << 5 | lane) << 4) + idx] = (_Float16)v;
}

// pack f32 weight (K x N row-major, or N x K if transB) into B panels
__global__ void packB_kernel(const float* __restrict__ B, _Float16* __restrict__ P,
                             int K, int N, int transB) {
  int nt = blockIdx.x;           // gridDim.x = nTiles
  int ks = blockIdx.y;           // gridDim.y = KS
  for (int e = threadIdx.x; e < 512; e += 256) {
    int lane = e >> 4, idx = e & 15;
    int kkl = ((lane >> 4) << 3) + (idx & 7) + ((idx >> 3) << 4);    // B layout: k = 8*half + s (s<8), 16 + 8*half + (s-8) (s>=8)
    int c  = (nt << 4) + (lane & 15);
    int kk = (ks << 5) + kkl;
    float v = 0.0f;
    if (c < N && kk < K)
      v = transB ? B[(size_t)c * K + kk] : B[(size_t)kk * N + c];
    P[(((size_t)nt * gridDim.y + ks) << 9) + e] = (_Float16)v;
  }
}

// =====================================================================
// Panelized WMMA GEMM:  C[M,N] (+)= epi(alpha*A@B + bias) [* rowScale]
// Optionally also emits C as A-side f16 panels (Cpan) for the next GEMM.
// Block 256 thr = 8 waves (2Mx4N); wave = 32x32 out (2x2 WMMA tiles);
// block tile 64x128, K chunk 64 (2 ksteps). LDS double-buffered, staged
// with global_load_async_to_lds_b128 (ASYNCcnt) when available.
// =====================================================================
__launch_bounds__(256)
__global__ void wmma_gemm_p(const _Float16* __restrict__ A16, const _Float16* __restrict__ B16,
                            float* __restrict__ C, _Float16* __restrict__ Cpan, int CpanKS,
                            int M, int N, int K,
                            const float* __restrict__ bias,
                            const float* __restrict__ rowScale, int rsStride,
                            int accumulate, int epilogue, float alpha) {
  __shared__ __align__(32) _Float16 Ab[2][8][512];    // 8 A panels / chunk (8KB) x2
  __shared__ __align__(32) _Float16 Bb[2][16][512];   // 16 B panels / chunk (16KB) x2

  const int tid   = threadIdx.x;
  const int lane  = tid & 31;
  const int wave  = tid >> 5;
  const int waveM = wave >> 2;          // 0..1
  const int waveN = wave & 3;           // 0..3
  const int g     = lane >> 4;
  const int lr    = lane & 15;

  const int KS     = (K + 31) >> 5;
  const int mTiles = (M + 15) >> 4;
  const int nTiles = (N + 15) >> 4;
  const int mt0    = blockIdx.y << 2;   // 4 row tiles per block
  const int nt0    = blockIdx.x << 3;   // 8 col tiles per block
  const int nChunks = (KS + 1) >> 1;

  CFrag acc[2][2];
#pragma unroll
  for (int tm = 0; tm < 2; ++tm)
#pragma unroll
    for (int tn = 0; tn < 2; ++tn)
      acc[tm][tn].v = (v8f){0.f,0.f,0.f,0.f,0.f,0.f,0.f,0.f};

  // stage one 64-K chunk (24 panels, 24KB) into buffer pbuf
  auto stage = [&](int chunk, int pbuf) {
    int ks0 = chunk << 1;
    for (int u = tid; u < 1536; u += 256) {          // 1536 x 16B units
      int panel = u >> 6, sub = u & 63;
      const _Float16* src; _Float16* dst; bool valid;
      if (panel < 8) {
        int mt = mt0 + (panel >> 1), kss = ks0 + (panel & 1);
        valid = (mt < mTiles) && (kss < KS);
        src = A16 + ((((size_t)mt * KS + kss) << 9) + (sub << 3));
        dst = &Ab[pbuf][panel][sub << 3];
      } else {
        int p = panel - 8;
        int nt = nt0 + (p >> 1), kss = ks0 + (p & 1);
        valid = (nt < nTiles) && (kss < KS);
        src = B16 + ((((size_t)nt * KS + kss) << 9) + (sub << 3));
        dst = &Bb[pbuf][p][sub << 3];
      }
      if (valid) {
#ifdef USE_ASYNC_LDS
        async_cp16(src, dst);
#else
        *(uint4*)dst = *(const uint4*)src;
#endif
      } else {
        *(uint4*)dst = (uint4){0u, 0u, 0u, 0u};
      }
    }
  };

  stage(0, 0);
  async_wait0();
  __syncthreads();

  int pb = 0;
  for (int c = 0; c < nChunks; ++c) {
    if (c + 1 < nChunks) stage(c + 1, pb ^ 1);
    // prefetch chunk c+2 panel heads (global_prefetch_b8)
    if (c + 2 < nChunks && tid < 24) {
      int ks2 = (c + 2) << 1;
      if (tid < 8) {
        int mt = mt0 + (tid >> 1);
        if (mt < mTiles) __builtin_prefetch(A16 + (((size_t)mt * KS + ks2 + (tid & 1)) << 9), 0, 0);
      } else {
        int p = tid - 8;
        int nt = nt0 + (p >> 1);
        if (nt < nTiles) __builtin_prefetch(B16 + (((size_t)nt * KS + ks2 + (p & 1)) << 9), 0, 0);
      }
    }
#pragma unroll
    for (int ks = 0; ks < 2; ++ks) {
      v16h aF[2], bF[2];
#pragma unroll
      for (int tm = 0; tm < 2; ++tm)
        aF[tm] = *(const v16h*)&Ab[pb][(((waveM << 1) + tm) << 1) + ks][lane << 4];
#pragma unroll
      for (int tn = 0; tn < 2; ++tn)
        bF[tn] = *(const v16h*)&Bb[pb][(((waveN << 1) + tn) << 1) + ks][lane << 4];
#pragma unroll
      for (int tm = 0; tm < 2; ++tm)
#pragma unroll
        for (int tn = 0; tn < 2; ++tn)
          acc[tm][tn].v = __builtin_amdgcn_wmma_f32_16x16x32_f16(
              false, aF[tm], false, bF[tn], (short)0, acc[tm][tn].v, false, false);
    }
    async_wait0();
    __syncthreads();
    pb ^= 1;
  }

  // epilogue (C/D layout: lanes 0-15 -> M=v, lanes 16-31 -> M=8+v)
#pragma unroll
  for (int tm = 0; tm < 2; ++tm) {
#pragma unroll
    for (int tn = 0; tn < 2; ++tn) {
#pragma unroll
      for (int vI = 0; vI < 8; ++vI) {
        int row = ((int)blockIdx.y << 6) + (waveM << 5) + (tm << 4) + (g << 3) + vI;
        int col = ((int)blockIdx.x << 7) + (waveN << 5) + (tn << 4) + lr;
        if (row < M && col < N) {
          float x = acc[tm][tn].f[vI] * alpha;
          if (bias) x += bias[col];
          if (epilogue == 1) {
            float x3 = x * x * x;
            x = 0.5f * x * (1.0f + tanhf(0.7978845608028654f * (x + 0.044715f * x3)));
          } else if (epilogue >= 2) {
            x = (x > 20.0f) ? x : log1pf(expf(x));
            if (epilogue == 3) x += 0.1f;
          }
          if (rowScale) x *= rowScale[(size_t)row * rsStride];
          if (Cpan) storeApanel(Cpan, CpanKS, row, col, x);
          if (C) {
            size_t idx = (size_t)row * N + col;
            if (accumulate) C[idx] += x; else C[idx] = x;
          }
        }
      }
    }
  }
}

// =====================================================================
// small helper kernels
// =====================================================================
__device__ __forceinline__ float block_reduce_sum(float v, float* sm) {
  int tid = threadIdx.x;
  sm[tid] = v; __syncthreads();
  for (int off = 128; off > 0; off >>= 1) {
    if (tid < off) sm[tid] += sm[tid + off];
    __syncthreads();
  }
  float r = sm[0]; __syncthreads();
  return r;
}

// h = emb[x]+pos, emitted directly as f16 A panels
__global__ void embed_kernel(const int* __restrict__ x, const float* __restrict__ emb,
                             const float* __restrict__ pos, _Float16* __restrict__ H16) {
  int tok = blockIdx.x;
  int t = tok % TD;
  int id = x[tok];
  for (int d = threadIdx.x; d < DD; d += blockDim.x) {
    float v = emb[(size_t)id * DD + d] + pos[(size_t)t * DD + d];
    storeApanel(H16, KS_D, tok, d, v);
  }
}

__global__ void pi_init_kernel(float* PI) {
  int i = blockIdx.x * blockDim.x + threadIdx.x;
  if (i < MROWS * NS) PI[i] = ((i % NS) == 2) ? 1.0f : 0.0f;
}

__global__ void zero_kernel(float* p, int n) {
  int i = blockIdx.x * blockDim.x + threadIdx.x;
  if (i < n) p[i] = 0.0f;
}

// K = softmax(trans.reshape(NS,NS), -1);  pi_ev = pi @ K
__global__ void trans_pi_kernel(const float* __restrict__ TR, const float* __restrict__ PI,
                                float* __restrict__ PIEV) {
  int tok = blockIdx.x * blockDim.x + threadIdx.x;
  if (tok >= MROWS) return;
  const float* tr = TR + (size_t)tok * (NS * NS);
  const float* pi = PI + (size_t)tok * NS;
  float Km[NS][NS];
  for (int s = 0; s < NS; ++s) {
    float mx = -1e30f;
    for (int r = 0; r < NS; ++r) mx = fmaxf(mx, tr[s * NS + r]);
    float sum = 0.0f;
    for (int r = 0; r < NS; ++r) { Km[s][r] = expf(tr[s * NS + r] - mx); sum += Km[s][r]; }
    float inv = 1.0f / sum;
    for (int r = 0; r < NS; ++r) Km[s][r] *= inv;
  }
  for (int r = 0; r < NS; ++r) {
    float a = 0.0f;
    for (int s = 0; s < NS; ++s) a += pi[s] * Km[s][r];
    PIEV[(size_t)tok * NS + r] = a;
  }
}

// pi_new = pi_ev * softmax(ev/0.5); normalize; top-2 mask; renormalize
__global__ void pi_update_kernel(const float* __restrict__ EVID, const float* __restrict__ PIEV,
                                 float* __restrict__ PI) {
  int tok = blockIdx.x * blockDim.x + threadIdx.x;
  if (tok >= MROWS) return;
  const float* ev = EVID + (size_t)tok * NS;
  float sm[NS], pn[NS];
  float mx = -1e30f;
  for (int j = 0; j < NS; ++j) mx = fmaxf(mx, ev[j] * 2.0f);
  float sum = 0.0f;
  for (int j = 0; j < NS; ++j) { sm[j] = expf(ev[j] * 2.0f - mx); sum += sm[j]; }
  float inv = 1.0f / sum;
  float tot = 0.0f;
  for (int j = 0; j < NS; ++j) { pn[j] = PIEV[(size_t)tok * NS + j] * sm[j] * inv; tot += pn[j]; }
  tot = fmaxf(tot, 1e-8f);
  for (int j = 0; j < NS; ++j) pn[j] /= tot;
  int i1 = 0;
  for (int j = 1; j < NS; ++j) if (pn[j] > pn[i1]) i1 = j;
  int i2 = -1;
  for (int j = 0; j < NS; ++j) if (j != i1 && (i2 < 0 || pn[j] > pn[i2])) i2 = j;
  float den = fmaxf(pn[i1] + pn[i2], 1e-8f);
  for (int j = 0; j < NS; ++j)
    PI[(size_t)tok * NS + j] = (j == i1) ? pn[i1] / den : ((j == i2) ? pn[i2] / den : 0.0f);
}

// windowed router attention: MSG = attn(q,k,v over s=1..4) * pi[:,3]
__launch_bounds__(256)
__global__ void router_kernel(const float* __restrict__ Q, const float* __restrict__ Kc,
                              const float* __restrict__ Vc, const float* __restrict__ PI,
                              float* __restrict__ MSG) {
  __shared__ float red[256];
  __shared__ float prob[WIN];
  int tok = blockIdx.x;
  int t = tok % TD;
  float part[WIN] = {0.f, 0.f, 0.f, 0.f};
  for (int d = threadIdx.x; d < DD; d += 256) {
    float q = Q[(size_t)tok * DD + d];
    for (int s = 1; s <= WIN; ++s)
      if (t >= s) part[s - 1] += q * Kc[(size_t)(tok - s) * DD + d];
  }
  float sc[WIN];
  for (int s = 0; s < WIN; ++s) sc[s] = block_reduce_sum(part[s], red);
  if (threadIdx.x == 0) {
    float mx = -1e30f;
    for (int s = 0; s < WIN; ++s) {
      sc[s] = (t >= s + 1) ? sc[s] * RSQRT_D : -1e9f;
      mx = fmaxf(mx, sc[s]);
    }
    float sum = 0.0f;
    for (int s = 0; s < WIN; ++s) { sc[s] = expf(sc[s] - mx); sum += sc[s]; }
    for (int s = 0; s < WIN; ++s) prob[s] = (t >= s + 1) ? sc[s] / sum : 0.0f;
  }
  __syncthreads();
  float g3 = PI[(size_t)tok * NS + 3];
  for (int d = threadIdx.x; d < DD; d += 256) {
    float o = 0.0f;
    for (int s = 1; s <= WIN; ++s)
      if (t >= s) o += prob[s - 1] * Vc[(size_t)(tok - s) * DD + d];
    MSG[(size_t)tok * DD + d] = o * g3;
  }
}

// memory read: MSG += 0.1*g3*softmax(mu.mem/sqrt(D))@mem; emit MSG f16 panels
__launch_bounds__(256)
__global__ void memctx_kernel(const float* __restrict__ MU, const float* __restrict__ MEM,
                              const float* __restrict__ PI, float* __restrict__ MSG,
                              _Float16* __restrict__ MSG16) {
  __shared__ float red[256];
  __shared__ float prob[NSL];
  int tok = blockIdx.x;
  int b = tok / TD;
  float part[NSL] = {0.f};
  for (int d = threadIdx.x; d < DD; d += 256) {
    float m = MU[(size_t)tok * DD + d];
    for (int k = 0; k < NSL; ++k)
      part[k] += m * MEM[((size_t)b * NSL + k) * DD + d];
  }
  for (int k = 0; k < NSL; ++k) {
    float tot = block_reduce_sum(part[k], red);
    if (threadIdx.x == 0) prob[k] = tot * RSQRT_D;
    __syncthreads();
  }
  if (threadIdx.x == 0) {
    float mx = -1e30f;
    for (int k = 0; k < NSL; ++k) mx = fmaxf(mx, prob[k]);
    float sum = 0.0f;
    for (int k = 0; k < NSL; ++k) { prob[k] = expf(prob[k] - mx); sum += prob[k]; }
    for (int k = 0; k < NSL; ++k) prob[k] /= sum;
  }
  __syncthreads();
  float g3 = PI[(size_t)tok * NS + 3];
  for (int d = threadIdx.x; d < DD; d += 256) {
    float o = 0.0f;
    for (int k = 0; k < NSL; ++k) o += prob[k] * MEM[((size_t)b * NSL + k) * DD + d];
    float v = MSG[(size_t)tok * DD + d] + 0.1f * o * g3;
    MSG[(size_t)tok * DD + d] = v;
    storeApanel(MSG16, KS_D, tok, d, v);
  }
}

// Bayesian fusion + stage_out injection; refreshes MU f32 and f16 panels
__global__ void bayes_kernel(float* __restrict__ MU, _Float16* __restrict__ MU16,
                             float* __restrict__ LAM,
                             const float* __restrict__ OBS, const float* __restrict__ OLAM,
                             const float* __restrict__ SO, const float* __restrict__ PI) {
  int i = blockIdx.x * blockDim.x + threadIdx.x;
  if (i >= MROWS * DD) return;
  int tok = i / DD;
  int d   = i - tok * DD;
  float g4 = PI[(size_t)tok * NS + 4];
  float ol = OLAM[i];
  float lam = LAM[i];
  float lamNew = lam + g4 * ol;
  float mu = (lam * MU[i] + g4 * ol * OBS[i]) / lamNew;
  mu += 0.1f * SO[i];
  MU[i]  = mu;
  LAM[i] = lamNew;
  storeApanel(MU16, KS_D, tok, d, mu);
}

// w_sc = softmax((mu@scr_key).mem/sqrt(D)) * g4
__launch_bounds__(256)
__global__ void wsc_kernel(const float* __restrict__ SK, const float* __restrict__ MEM,
                           const float* __restrict__ PI, float* __restrict__ WSC) {
  __shared__ float red[256];
  __shared__ float prob[NSL];
  int tok = blockIdx.x;
  int b = tok / TD;
  float part[NSL] = {0.f};
  for (int d = threadIdx.x; d < DD; d += 256) {
    float m = SK[(size_t)tok * DD + d];
    for (int k = 0; k < NSL; ++k)
      part[k] += m * MEM[((size_t)b * NSL + k) * DD + d];
  }
  for (int k = 0; k < NSL; ++k) {
    float tot = block_reduce_sum(part[k], red);
    if (threadIdx.x == 0) prob[k] = tot * RSQRT_D;
    __syncthreads();
  }
  if (threadIdx.x == 0) {
    float mx = -1e30f;
    for (int k = 0; k < NSL; ++k) mx = fmaxf(mx, prob[k]);
    float sum = 0.0f;
    for (int k = 0; k < NSL; ++k) { prob[k] = expf(prob[k] - mx); sum += prob[k]; }
    float g4 = PI[(size_t)tok * NS + 4];
    for (int k = 0; k < NSL; ++k) WSC[(size_t)tok * NSL + k] = (prob[k] / sum) * g4;
  }
}

// mem[b,k,:] += (1/T) * sum_t wsc[t,k] * sv[t,:]
__launch_bounds__(256)
__global__ void memupd_kernel(const float* __restrict__ WSC, const float* __restrict__ SV,
                              float* __restrict__ MEM) {
  int b = blockIdx.x / NSL;
  int k = blockIdx.x % NSL;
  for (int d = threadIdx.x; d < DD; d += 256) {
    float acc = 0.0f;
    for (int t = 0; t < TD; ++t) {
      int tok = b * TD + t;
      acc += WSC[(size_t)tok * NSL + k] * SV[(size_t)tok * DD + d];
    }
    MEM[((size_t)b * NSL + k) * DD + d] += acc * (1.0f / TD);
  }
}

// LayerNorm, emitted directly as f16 A panels for the logits GEMM
__launch_bounds__(256)
__global__ void ln_kernel(const float* __restrict__ MU, const float* __restrict__ gw,
                          const float* __restrict__ bw, _Float16* __restrict__ FIN16) {
  __shared__ float red[256];
  int tok = blockIdx.x;
  float s = 0.0f;
  for (int d = threadIdx.x; d < DD; d += 256) s += MU[(size_t)tok * DD + d];
  float mean = block_reduce_sum(s, red) / DD;
  float v = 0.0f;
  for (int d = threadIdx.x; d < DD; d += 256) {
    float dx = MU[(size_t)tok * DD + d] - mean;
    v += dx * dx;
  }
  float var = block_reduce_sum(v, red) / DD;
  float inv = rsqrtf(var + 1e-5f);
  for (int d = threadIdx.x; d < DD; d += 256) {
    float y = (MU[(size_t)tok * DD + d] - mean) * inv * gw[d] + bw[d];
    storeApanel(FIN16, KS_D, tok, d, y);
  }
}

// =====================================================================
// host-side orchestration
// =====================================================================
extern "C" void kernel_launch(void* const* d_in, const int* in_sizes, int n_in,
                              void* d_out, int out_size, void* d_ws, size_t ws_size,
                              hipStream_t stream) {
  (void)in_sizes; (void)n_in; (void)out_size; (void)ws_size;

  const int*   x       = (const int*)  d_in[0];
  const float* emb     = (const float*)d_in[1];
  const float* pos     = (const float*)d_in[2];
  const float* mu_w    = (const float*)d_in[3];
  const float* mu_b    = (const float*)d_in[4];
  const float* lam_w   = (const float*)d_in[5];
  const float* lam_b   = (const float*)d_in[6];
  const float* trans_w = (const float*)d_in[7];
  const float* trans_b = (const float*)d_in[8];
  const float* ev_w    = (const float*)d_in[9];
  const float* ev_b    = (const float*)d_in[10];
  const float* ffn_w1  = (const float*)d_in[11];
  const float* ffn_b1  = (const float*)d_in[12];
  const float* ffn_w2  = (const float*)d_in[13];
  const float* ffn_b2  = (const float*)d_in[14];
  const float* rq      = (const float*)d_in[15];
  const float* rk      = (const float*)d_in[16];
  const float* rv      = (const float*)d_in[17];
  const float* obs_w   = (const float*)d_in[18];
  const float* obs_b   = (const float*)d_in[19];
  const float* olam_w  = (const float*)d_in[20];
  const float* olam_b  = (const float*)d_in[21];
  const float* scr_key = (const float*)d_in[22];
  const float* scr_val = (const float*)d_in[23];
  const float* ln_g    = (const float*)d_in[24];
  const float* ln_b    = (const float*)d_in[25];

  char* base = (char*)d_ws;
  size_t off = 0;
  auto allocB = [&](size_t bytes) { void* p = base + off; off += (bytes + 1023) & ~(size_t)1023; return p; };
  auto panelBytes = [](int M, int K) {
    return (size_t)((M + 15) / 16) * ((K + 31) / 32) * 512 * sizeof(_Float16);
  };

  // f32 buffers
  float* MU    = (float*)allocB((size_t)MROWS * DD * 4);
  float* LAM   = (float*)allocB((size_t)MROWS * DD * 4);
  float* PI    = (float*)allocB((size_t)MROWS * NS * 4);
  float* PIEV  = (float*)allocB((size_t)MROWS * NS * 4);
  float* TRB   = (float*)allocB((size_t)MROWS * NS * NS * 4);
  float* EVID  = (float*)allocB((size_t)MROWS * NS * 4);
  float* SO    = (float*)allocB((size_t)MROWS * DD * 4);
  float* Qb    = (float*)allocB((size_t)MROWS * DD * 4);
  float* Kb    = (float*)allocB((size_t)MROWS * DD * 4);
  float* Vb    = (float*)allocB((size_t)MROWS * DD * 4);
  float* MSG   = (float*)allocB((size_t)MROWS * DD * 4);
  float* OBS   = (float*)allocB((size_t)MROWS * DD * 4);
  float* OLAMB = (float*)allocB((size_t)MROWS * DD * 4);
  float* SKb   = (float*)allocB((size_t)MROWS * DD * 4);
  float* SVb   = (float*)allocB((size_t)MROWS * DD * 4);
  float* WSC   = (float*)allocB((size_t)MROWS * NSL * 4);
  float* MEM   = (float*)allocB((size_t)BD * NSL * DD * 4);

  // f16 activation panels
  _Float16* H16   = (_Float16*)allocB(panelBytes(MROWS, DD));
  _Float16* MU16  = (_Float16*)allocB(panelBytes(MROWS, DD));
  _Float16* HID16 = (_Float16*)allocB(panelBytes(MROWS, FFD));
  _Float16* MSG16 = (_Float16*)allocB(panelBytes(MROWS, DD));
  _Float16* FIN16 = (_Float16*)allocB(panelBytes(MROWS, DD));

  // f16 weight panels (B-side); sized by (N tiles, K steps)
  auto wpanelBytes = [](int K, int N) {
    return (size_t)((N + 15) / 16) * ((K + 31) / 32) * 512 * sizeof(_Float16);
  };
  _Float16* muW16   = (_Float16*)allocB(wpanelBytes(DD, DD));
  _Float16* lamW16  = (_Float16*)allocB(wpanelBytes(DD, DD));
  _Float16* trW16   = (_Float16*)allocB(wpanelBytes(DD, NS * NS));
  _Float16* evW16   = (_Float16*)allocB(wpanelBytes(DD, NS));
  _Float16* rqW16   = (_Float16*)allocB(wpanelBytes(DD, DD));
  _Float16* rkW16   = (_Float16*)allocB(wpanelBytes(DD, DD));
  _Float16* rvW16   = (_Float16*)allocB(wpanelBytes(DD, DD));
  _Float16* obsW16  = (_Float16*)allocB(wpanelBytes(DD, DD));
  _Float16* olamW16 = (_Float16*)allocB(wpanelBytes(DD, DD));
  _Float16* skW16   = (_Float16*)allocB(wpanelBytes(DD, DD));
  _Float16* svW16   = (_Float16*)allocB(wpanelBytes(DD, DD));
  _Float16* f1W16   = (_Float16*)allocB(NS * wpanelBytes(DD, FFD));
  _Float16* f2W16   = (_Float16*)allocB(NS * wpanelBytes(FFD, DD));
  _Float16* embW16  = (_Float16*)allocB(wpanelBytes(DD, VD));
  const size_t f1Stride = wpanelBytes(DD, FFD) / sizeof(_Float16);
  const size_t f2Stride = wpanelBytes(FFD, DD) / sizeof(_Float16);

  auto packB = [&](const float* W, _Float16* P, int K, int N, int transB) {
    dim3 grid((N + 15) / 16, (K + 31) / 32);
    packB_kernel<<<grid, 256, 0, stream>>>(W, P, K, N, transB);
  };
  auto gemm = [&](const _Float16* A16, const _Float16* B16, float* C,
                  _Float16* Cpan, int CpanKS, int m, int n, int k,
                  const float* bias, const float* rowScale, int rsStride,
                  int accumulate, int epi, float alpha) {
    dim3 grid((n + 127) / 128, (m + 63) / 64);
    wmma_gemm_p<<<grid, 256, 0, stream>>>(A16, B16, C, Cpan, CpanKS, m, n, k,
                                          bias, rowScale, rsStride, accumulate, epi, alpha);
  };

  // ---- pack all weights into B-side fragment panels (once per launch) ----
  packB(mu_w,   muW16,  DD, DD, 0);
  packB(lam_w,  lamW16, DD, DD, 0);
  packB(trans_w, trW16, DD, NS * NS, 0);
  packB(ev_w,   evW16,  DD, NS, 0);
  packB(rq, rqW16, DD, DD, 0);
  packB(rk, rkW16, DD, DD, 0);
  packB(rv, rvW16, DD, DD, 0);
  packB(obs_w,  obsW16,  DD, DD, 0);
  packB(olam_w, olamW16, DD, DD, 0);
  packB(scr_key, skW16, DD, DD, 0);
  packB(scr_val, svW16, DD, DD, 0);
  for (int s = 0; s < NS; ++s) {
    packB(ffn_w1 + (size_t)s * DD * FFD, f1W16 + (size_t)s * f1Stride, DD, FFD, 0);
    packB(ffn_w2 + (size_t)s * FFD * DD, f2W16 + (size_t)s * f2Stride, FFD, DD, 0);
  }
  packB(emb, embW16, DD, VD, 1);   // emb^T for logits

  // ---- embedding + initial heads ----
  embed_kernel<<<MROWS, 256, 0, stream>>>(x, emb, pos, H16);
  gemm(H16, muW16,  MU,  MU16, KS_D, MROWS, DD, DD, mu_b,  nullptr, 0, 0, 0, 1.0f);
  gemm(H16, lamW16, LAM, nullptr, 0,  MROWS, DD, DD, lam_b, nullptr, 0, 0, 3, 1.0f);
  pi_init_kernel<<<(MROWS * NS + 255) / 256, 256, 0, stream>>>(PI);
  zero_kernel<<<(BD * NSL * DD + 255) / 256, 256, 0, stream>>>(MEM, BD * NSL * DD);

  // ---- iterative refinement ----
  for (int step = 0; step < NSTEPS; ++step) {
    // transition + pi evolution (uses OLD pi)
    gemm(MU16, trW16, TRB, nullptr, 0, MROWS, NS * NS, DD, trans_b, nullptr, 0, 0, 0, 1.0f);
    trans_pi_kernel<<<(MROWS + 255) / 256, 256, 0, stream>>>(TRB, PI, PIEV);

    // stage FFNs, gated by OLD pi, accumulated into SO (hidden stays f16-only)
    for (int s = 0; s < NS; ++s) {
      gemm(MU16, f1W16 + (size_t)s * f1Stride, nullptr, HID16, KS_FF,
           MROWS, FFD, DD, ffn_b1 + (size_t)s * FFD, nullptr, 0, 0, 1 /*gelu*/, 1.0f);
      gemm(HID16, f2W16 + (size_t)s * f2Stride, SO, nullptr, 0,
           MROWS, DD, FFD, ffn_b2 + (size_t)s * DD, PI + s, NS, (s > 0) ? 1 : 0, 0, 1.0f);
    }

    // evidence + pi update (writes NEW pi)
    gemm(MU16, evW16, EVID, nullptr, 0, MROWS, NS, DD, ev_b, nullptr, 0, 0, 0, 1.0f);
    pi_update_kernel<<<(MROWS + 255) / 256, 256, 0, stream>>>(EVID, PIEV, PI);

    // router attention (NEW pi gate g3)
    gemm(MU16, rqW16, Qb, nullptr, 0, MROWS, DD, DD, nullptr, nullptr, 0, 0, 0, 1.0f);
    gemm(MU16, rkW16, Kb, nullptr, 0, MROWS, DD, DD, nullptr, nullptr, 0, 0, 0, 1.0f);
    gemm(MU16, rvW16, Vb, nullptr, 0, MROWS, DD, DD, nullptr, nullptr, 0, 0, 0, 1.0f);
    router_kernel<<<MROWS, 256, 0, stream>>>(Qb, Kb, Vb, PI, MSG);
    memctx_kernel<<<MROWS, 256, 0, stream>>>(MU, MEM, PI, MSG, MSG16);

    // observation heads
    gemm(MSG16, obsW16,  OBS,   nullptr, 0, MROWS, DD, DD, obs_b,  nullptr, 0, 0, 0, 1.0f);
    gemm(MSG16, olamW16, OLAMB, nullptr, 0, MROWS, DD, DD, olam_b, nullptr, 0, 0, 2, 1.0f);

    // Bayesian fusion + stage_out (refreshes MU f32 + MU16 panels)
    bayes_kernel<<<(MROWS * DD + 255) / 256, 256, 0, stream>>>(MU, MU16, LAM, OBS, OLAMB, SO, PI);

    // scratch-memory write (uses updated mu)
    gemm(MU16, skW16, SKb, nullptr, 0, MROWS, DD, DD, nullptr, nullptr, 0, 0, 0, 1.0f);
    gemm(MU16, svW16, SVb, nullptr, 0, MROWS, DD, DD, nullptr, nullptr, 0, 0, 0, 1.0f);
    wsc_kernel<<<MROWS, 256, 0, stream>>>(SKb, MEM, PI, WSC);
    memupd_kernel<<<BD * NSL, 256, 0, stream>>>(WSC, SVb, MEM);
  }

  // ---- final LN + logits ----
  ln_kernel<<<MROWS, 256, 0, stream>>>(MU, ln_g, ln_b, FIN16);
  gemm(FIN16, embW16, (float*)d_out, nullptr, 0, MROWS, VD, DD,
       nullptr, nullptr, 0, 0, 0, RSQRT_D);
}
